// ReconstructionHead_44238163149316
// MI455X (gfx1250) — matrix operand order repaired and checked by reading, not compiled, not run on hardware
//
#include <hip/hip_runtime.h>

// MI455X (gfx1250) fused kernel:
//   out[b,t] = relu(LN(x[b,t,:] @ W1^T + b1)) . Wout[t,:] + bout[t]
// GEMM in f16 WMMA (f32 accum); 64 rows/WG, wave = 64 rows x 64 cols
// (4 M-tiles x 4 N-tiles). W1/Wout pre-converted to f16 (L2-resident);
// x streamed with non-temporal loads (zero reuse -> keep L2 for weights).

#define DIM     512
#define EPS_LN  1e-5f
#define ROWS_WG 64          // rows (b,t pairs) per workgroup
#define MT      4           // 16-row M tiles per wave
#define NT      4           // 16-col N tiles per wave (64 cols)

typedef __attribute__((ext_vector_type(16))) _Float16 v16h;
typedef __attribute__((ext_vector_type(8)))  _Float16 v8h;
typedef __attribute__((ext_vector_type(8)))  float    v8f;
typedef __attribute__((ext_vector_type(4)))  float    v4f;

// ---------- pre-pass: f32 -> f16 weight conversion into workspace ----------
__global__ void convert_f32_to_f16(const float* __restrict__ src,
                                   _Float16* __restrict__ dst, int n) {
    int i = blockIdx.x * blockDim.x + threadIdx.x;
    if (i < n) dst[i] = (_Float16)src[i];
}

// ---------- fused GEMM + LayerNorm + ReLU + row-dot ----------
__global__ __launch_bounds__(256)
void fused_head(const float* __restrict__ x,        // (256, 513, 512) f32
                const _Float16* __restrict__ W1h,   // (512, 512) f16, row-major (e,d)
                const float* __restrict__ b1,
                const float* __restrict__ gamma,
                const float* __restrict__ beta,
                const _Float16* __restrict__ Wouth, // (512, 512) f16, row-major (t,d)
                const float* __restrict__ bout,
                float* __restrict__ out)            // (256*512,) f32 flat
{
    // x tile stored in WMMA A-fragment order: [mtile(4)][kt(16)][lane(32)][16 halfs]
    // = 65536 B. Reduction scratch aliases this region after the GEMM.
    __shared__ alignas(32) _Float16 xth[MT * 16 * 32 * 16];

    const int tid  = threadIdx.x;
    const int lane = tid & 31;
    const int wv   = tid >> 5;       // wave 0..7 -> column group
    const int lid  = lane & 15;
    const int half = lane >> 4;      // 0 / 1
    const int rb   = blockIdx.x * ROWS_WG;   // global flat row base (b*512 + t)
    const int bidx = rb >> 9;                // batch index
    const int t0   = rb & 511;               // t base (block never crosses batch)
    const int cb   = wv * 64;                // this wave's column base

    // ---- cooperative load: 64 rows x 512 f32 -> f16 LDS tile in fragment order ----
    // Non-temporal: x has zero reuse; do not displace weights from L2.
    {
        const int row  = tid >> 2;           // 0..63
        const int mt   = row >> 4;
        const int M    = row & 15;
        const float* src = x + (size_t)(bidx * 513 + t0 + row) * DIM;
        #pragma unroll
        for (int j = 0; j < 16; ++j) {
            const int k8   = (tid & 3) * 16 + j;   // 8-half chunk index 0..63
            const int kt   = k8 >> 2;
            const int sub  = k8 & 3;
            const int lnT  = M + (sub >> 1) * 16;  // destination lane of fragment
            const int eoff = (sub & 1) * 8;        // element offset inside v16h
            v4f f0 = __builtin_nontemporal_load((const v4f*)(src + k8 * 8));
            v4f f1 = __builtin_nontemporal_load((const v4f*)(src + k8 * 8 + 4));
            v8h hseg;
            #pragma unroll
            for (int e = 0; e < 4; ++e) {
                hseg[e]     = (_Float16)f0[e];
                hseg[e + 4] = (_Float16)f1[e];
            }
            *(v8h*)(xth + ((size_t)((mt * 16 + kt) * 32 + lnT) * 16 + eoff)) = hseg;
        }
    }
    __syncthreads();

    // ---- GEMM: wave = 64 rows x 64 cols ; K streamed 32 at a time ----
    v8f acc[MT][NT];
    #pragma unroll
    for (int m = 0; m < MT; ++m)
        #pragma unroll
        for (int n = 0; n < NT; ++n) { v8f z = {}; acc[m][n] = z; }

    for (int kt = 0; kt < 16; ++kt) {
        v16h a[MT];
        #pragma unroll
        for (int m = 0; m < MT; ++m)
            a[m] = *(const v16h*)(xth + (size_t)((m * 16 + kt) * 32 + lane) * 16);
        #pragma unroll
        for (int n = 0; n < NT; ++n) {
            const _Float16* wrow =
                W1h + (size_t)(cb + n * 16 + lid) * DIM + kt * 32 + half * 16;
            v16h bf = *(const v16h*)wrow;
            #pragma unroll
            for (int m = 0; m < MT; ++m)
                acc[m][n] = __builtin_amdgcn_wmma_f32_16x16x32_f16(
                    false, a[m], false, bf, (short)0, acc[m][n], false, false);
        }
    }

    // ---- add b1 (column-dependent, must precede LN stats) ----
    float b1v[NT], gv[NT], bev[NT];
    #pragma unroll
    for (int n = 0; n < NT; ++n) {
        const int col = cb + n * 16 + lid;
        b1v[n] = b1[col]; gv[n] = gamma[col]; bev[n] = beta[col];
    }
    #pragma unroll
    for (int m = 0; m < MT; ++m)
        #pragma unroll
        for (int n = 0; n < NT; ++n)
            #pragma unroll
            for (int v = 0; v < 8; ++v)
                acc[m][n][v] += b1v[n];

    // ---- x tile is dead: reuse its LDS for reductions ----
    __syncthreads();
    float* sred  = (float*)xth;              // [8 waves][64 rows][2]  (sum, sumsq)
    float* pred  = sred + 8 * 64 * 2;        // [8 waves][64 rows]     (dot partials)
    float* mstat = pred + 8 * 64;            // [64]  mean
    float* rstat = mstat + 64;               // [64]  rstd

    // per-row sum / sumsq partials for this wave's 64 columns
    #pragma unroll
    for (int m = 0; m < MT; ++m) {
        float s[8], q[8];
        #pragma unroll
        for (int v = 0; v < 8; ++v) { s[v] = 0.f; q[v] = 0.f; }
        #pragma unroll
        for (int n = 0; n < NT; ++n)
            #pragma unroll
            for (int v = 0; v < 8; ++v) {
                const float h = acc[m][n][v];
                s[v] += h; q[v] += h * h;
            }
        #pragma unroll
        for (int v = 0; v < 8; ++v) {
            #pragma unroll
            for (int off = 1; off < 16; off <<= 1) {
                s[v] += __shfl_xor(s[v], off, 32);
                q[v] += __shfl_xor(q[v], off, 32);
            }
        }
        if (lid == 0) {
            #pragma unroll
            for (int v = 0; v < 8; ++v) {
                const int row = m * 16 + half * 8 + v;
                sred[(wv * 64 + row) * 2 + 0] = s[v];
                sred[(wv * 64 + row) * 2 + 1] = q[v];
            }
        }
    }
    __syncthreads();

    // finalize LN stats, one thread per row
    if (tid < ROWS_WG) {
        float ss = 0.f, qq = 0.f;
        #pragma unroll
        for (int w = 0; w < 8; ++w) {
            ss += sred[(w * 64 + tid) * 2 + 0];
            qq += sred[(w * 64 + tid) * 2 + 1];
        }
        const float m  = ss * (1.0f / 512.0f);
        const float va = qq * (1.0f / 512.0f) - m * m;
        mstat[tid] = m;
        rstat[tid] = rsqrtf(va + EPS_LN);
    }
    __syncthreads();

    // ---- normalize, ReLU, dot with Wout[t,:] (per M-tile to cap live regs) ----
    #pragma unroll
    for (int m = 0; m < MT; ++m) {
        float mean[8], rstd[8];
        int   trow[8];
        #pragma unroll
        for (int v = 0; v < 8; ++v) {
            const int row = m * 16 + half * 8 + v;
            mean[v] = mstat[row];
            rstd[v] = rstat[row];
            trow[v] = t0 + row;
        }
        float p[8];
        #pragma unroll
        for (int v = 0; v < 8; ++v) p[v] = 0.f;
        #pragma unroll
        for (int n = 0; n < NT; ++n) {
            const int col = cb + n * 16 + lid;
            #pragma unroll
            for (int v = 0; v < 8; ++v) {
                float hn = (acc[m][n][v] - mean[v]) * rstd[v] * gv[n] + bev[n];
                hn = fmaxf(hn, 0.f);
                p[v] += hn * (float)Wouth[(size_t)trow[v] * DIM + col];
            }
        }
        #pragma unroll
        for (int v = 0; v < 8; ++v) {
            #pragma unroll
            for (int off = 1; off < 16; off <<= 1)
                p[v] += __shfl_xor(p[v], off, 32);
        }
        if (lid == 0) {
            #pragma unroll
            for (int v = 0; v < 8; ++v)
                pred[wv * 64 + (m * 16 + half * 8 + v)] = p[v];
        }
    }
    __syncthreads();

    if (tid < ROWS_WG) {
        float o = 0.f;
        #pragma unroll
        for (int w = 0; w < 8; ++w) o += pred[w * 64 + tid];
        __builtin_nontemporal_store(o + bout[(rb + tid) & 511], out + rb + tid);
    }
}

extern "C" void kernel_launch(void* const* d_in, const int* in_sizes, int n_in,
                              void* d_out, int out_size, void* d_ws, size_t ws_size,
                              hipStream_t stream) {
    (void)in_sizes; (void)n_in; (void)out_size; (void)ws_size;
    const float* x     = (const float*)d_in[0];
    const float* W1    = (const float*)d_in[1];
    const float* b1    = (const float*)d_in[2];
    const float* gamma = (const float*)d_in[3];
    const float* beta  = (const float*)d_in[4];
    const float* Wout  = (const float*)d_in[5];
    const float* bout  = (const float*)d_in[6];
    float* out = (float*)d_out;

    _Float16* W1h   = (_Float16*)d_ws;                 // 512 KB
    _Float16* Wouth = W1h + (size_t)DIM * DIM;         // 512 KB

    convert_f32_to_f16<<<(DIM * DIM) / 256, 256, 0, stream>>>(W1,   W1h,   DIM * DIM);
    convert_f32_to_f16<<<(DIM * DIM) / 256, 256, 0, stream>>>(Wout, Wouth, DIM * DIM);

    const int total_rows = 256 * 512;              // B * T
    fused_head<<<total_rows / ROWS_WG, 256, 0, stream>>>(
        x, W1h, b1, gamma, beta, Wouth, bout, out);
}
